// MultiHeadAttention_81449759802311
// MI455X (gfx1250) — matrix operand order repaired and checked by reading, not compiled, not run on hardware
//
#include <hip/hip_runtime.h>

// ---------------------------------------------------------------------------
// MHA forward for MI455X (gfx1250): bf16 WMMA GEMMs + flash attention.
// B=2, S=2048, D=1024, H=16, DK=64
// ---------------------------------------------------------------------------

typedef __bf16 bf16;
typedef __attribute__((ext_vector_type(16))) __bf16 v16bf;
typedef __attribute__((ext_vector_type(8)))  __bf16 v8bf;
typedef __attribute__((ext_vector_type(8)))  float  v8f;
typedef int v4i_vs __attribute__((vector_size(16)));   // matches builtin param

union FragBf { v16bf v; v8bf h[2]; };

__device__ __forceinline__ v8f wmma_bf16(v16bf a, v16bf b, v8f c) {
  return __builtin_amdgcn_wmma_f32_16x16x32_bf16(
      false, a, false, b, (short)0, c, false, false);
}

#define NEG_INF (-__builtin_inff())

// --- gfx1250 async global->LDS path (guarded: falls back to sync copies) ---
#if __has_builtin(__builtin_amdgcn_global_load_async_to_lds_b128)
#define HAVE_ASYNC_LOAD 1
#define ASYNC_LOAD_B128(g, l)                                          \
  __builtin_amdgcn_global_load_async_to_lds_b128(                      \
      (__attribute__((address_space(1))) v4i_vs*)(g),                  \
      (__attribute__((address_space(3))) v4i_vs*)(l), 0, 0)
#else
#define HAVE_ASYNC_LOAD 0
#endif

#if __has_builtin(__builtin_amdgcn_s_wait_asynccnt)
#define WAIT_ASYNC(n) __builtin_amdgcn_s_wait_asynccnt(n)
#else
#define WAIT_ASYNC(n) asm volatile("s_wait_asynccnt %0" ::"i"(n))
#endif

// ---------------------------------------------------------------------------
// Kernel 0: fp32 -> bf16 cast (x staged once for 3 projections)
// ---------------------------------------------------------------------------
__global__ void cast_f32_bf16(const float* __restrict__ in,
                              bf16* __restrict__ out, int n4) {
  int i = blockIdx.x * blockDim.x + threadIdx.x;
  if (i < n4) {
    float4 f = ((const float4*)in)[i];
    bf16* o = out + (size_t)i * 4;
    o[0] = (bf16)f.x; o[1] = (bf16)f.y; o[2] = (bf16)f.z; o[3] = (bf16)f.w;
  }
}

// ---------------------------------------------------------------------------
// Kernel 0b: weight prep  WT[n][k] = (bf16)W[k][n]   (tiled LDS transpose)
// ---------------------------------------------------------------------------
__global__ __launch_bounds__(256)
void transpose_cast(const float* __restrict__ W, bf16* __restrict__ WT,
                    int K, int N) {
  __shared__ float tile[32][33];
  int bn = blockIdx.x * 32;            // n base
  int bk = blockIdx.y * 32;            // k base
  int tx = threadIdx.x & 31;
  int ty = threadIdx.x >> 5;           // 0..7
#pragma unroll
  for (int i = 0; i < 4; ++i)
    tile[ty + 8 * i][tx] = W[(size_t)(bk + ty + 8 * i) * N + bn + tx];
  __syncthreads();
#pragma unroll
  for (int i = 0; i < 4; ++i)
    WT[(size_t)(bn + ty + 8 * i) * K + bk + tx] = (bf16)tile[tx][ty + 8 * i];
}

// ---------------------------------------------------------------------------
// Kernel 1: GEMM  out = A(bf16, MxK) @ WT^T + bias, optional RoPE.
// WT is the pre-transposed weight (N x K, bf16) so both LDS tiles stage as
// contiguous 16B chunks (async-to-LDS friendly). Double-buffered K loop.
// Block tile 128x128, 8 waves, each wave 32x64 (2x4 wmma subtiles), K-step 32.
// ---------------------------------------------------------------------------
#define AS_STR 40   // bf16 elements per LDS row (32 used + pad, keeps 16B align)
#define BS_STR 40

__global__ __launch_bounds__(256)
void gemm_bias_rope(const bf16* __restrict__ A, const bf16* __restrict__ WT,
                    const float* __restrict__ bias,
                    bf16* __restrict__ outb, float* __restrict__ outf,
                    int M, int K, int N, int S, int rope) {
  __shared__ __attribute__((aligned(16))) bf16 As[2][128 * AS_STR];
  __shared__ __attribute__((aligned(16))) bf16 Bs[2][128 * BS_STR];

  const int tid   = threadIdx.x;
  const int wave  = tid >> 5;
  const int lane  = tid & 31;
  const int hl    = lane & 15;
  const int hi    = lane >> 4;
  const int waveM = wave & 3;
  const int waveN = wave >> 2;
  const int rowBase = blockIdx.y * 128;
  const int colBase = blockIdx.x * 128;

  v8f acc[2][4];
#pragma unroll
  for (int mi = 0; mi < 2; ++mi)
#pragma unroll
    for (int ni = 0; ni < 4; ++ni)
#pragma unroll
      for (int r = 0; r < 8; ++r) acc[mi][ni][r] = 0.0f;

  // stage one 128x32 A tile + 128x32 WT tile into LDS buffer `buf`
  auto stageTile = [&](int buf, int k0) {
#pragma unroll
    for (int it = 0; it < 2; ++it) {
      int chunk = tid + it * 256;      // 0..511
      int row = chunk >> 2;            // 0..127
      int c8  = chunk & 3;             // 8-element column group
      const bf16* gA = &A [(size_t)(rowBase + row) * K + k0 + c8 * 8];
      const bf16* gB = &WT[(size_t)(colBase + row) * K + k0 + c8 * 8];
      bf16* lA = &As[buf][row * AS_STR + c8 * 8];
      bf16* lB = &Bs[buf][row * BS_STR + c8 * 8];
#if HAVE_ASYNC_LOAD
      ASYNC_LOAD_B128(gA, lA);
      ASYNC_LOAD_B128(gB, lB);
#else
      *(v8bf*)lA = *(const v8bf*)gA;
      *(v8bf*)lB = *(const v8bf*)gB;
#endif
    }
  };

  stageTile(0, 0);
#if HAVE_ASYNC_LOAD
  WAIT_ASYNC(0);
#endif
  __syncthreads();

  const int nk = K >> 5;
  for (int kt = 0; kt < nk; ++kt) {
    int cur = kt & 1;
    if (kt + 1 < nk) stageTile(cur ^ 1, (kt + 1) << 5);

    v16bf af[2];
#pragma unroll
    for (int mi = 0; mi < 2; ++mi) {
      int row = waveM * 32 + mi * 16 + hl;
      int kb  = hi ? 8 : 0;
      FragBf f;
      f.h[0] = *(const v8bf*)&As[cur][row * AS_STR + kb];
      f.h[1] = *(const v8bf*)&As[cur][row * AS_STR + kb + 16];
      af[mi] = f.v;
    }
    v16bf bfm[4];
#pragma unroll
    for (int ni = 0; ni < 4; ++ni) {
      int col = waveN * 64 + ni * 16 + hl;
      int kb  = hi ? 8 : 0;
      FragBf f;
      f.h[0] = *(const v8bf*)&Bs[cur][col * BS_STR + kb];
      f.h[1] = *(const v8bf*)&Bs[cur][col * BS_STR + kb + 16];
      bfm[ni] = f.v;
    }
#pragma unroll
    for (int mi = 0; mi < 2; ++mi)
#pragma unroll
      for (int ni = 0; ni < 4; ++ni)
        acc[mi][ni] = wmma_bf16(af[mi], bfm[ni], acc[mi][ni]);

#if HAVE_ASYNC_LOAD
    WAIT_ASYNC(0);
#endif
    __syncthreads();
  }

  // --- epilogue: bias, RoPE (on 64-wide heads), store. S is a power of two.
#pragma unroll
  for (int mi = 0; mi < 2; ++mi) {
#pragma unroll
    for (int ni = 0; ni < 4; ++ni) {
      float bv = bias[colBase + waveN * 64 + ni * 16 + hl];
#pragma unroll
      for (int r = 0; r < 8; ++r) acc[mi][ni][r] += bv;
    }
    if (rope) {
#pragma unroll
      for (int ni = 0; ni < 2; ++ni) {
        int j = ni * 16 + hl;                           // dk in [0,32)
        float theta = __expf(-0.28782314f * (float)j);  // 10000^(-j/32)
#pragma unroll
        for (int r = 0; r < 8; ++r) {
          int row = rowBase + waveM * 32 + mi * 16 + r + hi * 8;
          int s = row & (S - 1);
          float ang = (float)s * theta;
          float c = __cosf(ang), sn = __sinf(ang);
          float x1 = acc[mi][ni][r], x2 = acc[mi][ni + 2][r];
          acc[mi][ni][r]     = x1 * c - x2 * sn;
          acc[mi][ni + 2][r] = x2 * c + x1 * sn;
        }
      }
    }
#pragma unroll
    for (int ni = 0; ni < 4; ++ni) {
      int n = colBase + waveN * 64 + ni * 16 + hl;
#pragma unroll
      for (int r = 0; r < 8; ++r) {
        int row = rowBase + waveM * 32 + mi * 16 + r + hi * 8;
        if (outf) outf[(size_t)row * N + n] = acc[mi][ni][r];
        else      outb[(size_t)row * N + n] = (bf16)acc[mi][ni][r];
      }
    }
  }
}

// ---------------------------------------------------------------------------
// Kernel 2: causal flash attention, one wave = 16 queries, key tiles of 32.
// Q/K/V laid out (B, S, H*64) bf16. Output same layout, bf16.
// ---------------------------------------------------------------------------
#define VS_STR 40
#define PS_STR 40

__global__ __launch_bounds__(256)
void attn(const bf16* __restrict__ Q, const bf16* __restrict__ Kb,
          const bf16* __restrict__ Vb, bf16* __restrict__ O,
          int Bn, int S, int Hn, int D) {
  __shared__ __attribute__((aligned(16))) bf16 Vs[8][64 * VS_STR];
  __shared__ __attribute__((aligned(16))) bf16 Ps[8][16 * PS_STR];

  const int tid  = threadIdx.x;
  const int wave = tid >> 5;
  const int lane = tid & 31;
  const int hl   = lane & 15;
  const int hi   = lane >> 4;
  const int b = blockIdx.z, h = blockIdx.y;
  const int q0 = blockIdx.x * 128 + wave * 16;

  bf16* myVs = Vs[wave];
  bf16* myPs = Ps[wave];

  const bf16* qrow = Q + ((size_t)(b * S + q0 + hl)) * D + h * 64;
  v16bf aq[2];
#pragma unroll
  for (int kk = 0; kk < 2; ++kk) {
    int kb = kk * 32 + (hi ? 8 : 0);
    FragBf f;
    f.h[0] = *(const v8bf*)&qrow[kb];
    f.h[1] = *(const v8bf*)&qrow[kb + 16];
    aq[kk] = f.v;
  }

  float mrow[8], lrow[8];
  v8f oacc[4];
#pragma unroll
  for (int r = 0; r < 8; ++r) { mrow[r] = NEG_INF; lrow[r] = 0.0f; }
#pragma unroll
  for (int ni = 0; ni < 4; ++ni)
#pragma unroll
    for (int r = 0; r < 8; ++r) oacc[ni][r] = 0.0f;

  const int nkt = (q0 + 16 + 31) >> 5;
  for (int kt = 0; kt < nkt; ++kt) {
    { // stage V tile (32 keys x 64 dk) transposed: Vs[dk][key]
      int key = kt * 32 + lane;
      if (key > q0 + 15) key = q0 + 15;
      const bf16* vrow = Vb + ((size_t)(b * S + key)) * D + h * 64;
#pragma unroll
      for (int c = 0; c < 8; ++c) {
        v8bf vv = *(const v8bf*)&vrow[c * 8];
#pragma unroll
        for (int e = 0; e < 8; ++e)
          myVs[(c * 8 + e) * VS_STR + lane] = vv[e];
      }
    }
    v16bf bk[2][2];
#pragma unroll
    for (int j = 0; j < 2; ++j) {
      int key = kt * 32 + j * 16 + hl;
      if (key > q0 + 15) key = q0 + 15;
      const bf16* krow = Kb + ((size_t)(b * S + key)) * D + h * 64;
#pragma unroll
      for (int kk = 0; kk < 2; ++kk) {
        int kb = kk * 32 + (hi ? 8 : 0);
        FragBf f;
        f.h[0] = *(const v8bf*)&krow[kb];
        f.h[1] = *(const v8bf*)&krow[kb + 16];
        bk[j][kk] = f.v;
      }
    }
    v8f sc[2];
#pragma unroll
    for (int j = 0; j < 2; ++j) {
      v8f z;
#pragma unroll
      for (int r = 0; r < 8; ++r) z[r] = 0.0f;
      z = wmma_bf16(aq[0], bk[j][0], z);
      sc[j] = wmma_bf16(aq[1], bk[j][1], z);
    }
    float rmax[8];
#pragma unroll
    for (int r = 0; r < 8; ++r) rmax[r] = NEG_INF;
#pragma unroll
    for (int j = 0; j < 2; ++j) {
      int keyb = kt * 32 + j * 16 + hl;
#pragma unroll
      for (int r = 0; r < 8; ++r) {
        float v = sc[j][r] * 0.125f;
        int qq = q0 + r + hi * 8;
        v = (keyb > qq) ? NEG_INF : v;
        sc[j][r] = v;
        rmax[r] = fmaxf(rmax[r], v);
      }
    }
#pragma unroll
    for (int r = 0; r < 8; ++r)
#pragma unroll
      for (int m = 1; m < 16; m <<= 1)
        rmax[r] = fmaxf(rmax[r], __shfl_xor(rmax[r], m, 32));
#pragma unroll
    for (int r = 0; r < 8; ++r) {
      float mnew = fmaxf(mrow[r], rmax[r]);
      float corr = __expf(mrow[r] - mnew);
      lrow[r] *= corr;
      mrow[r] = mnew;
#pragma unroll
      for (int ni = 0; ni < 4; ++ni) oacc[ni][r] *= corr;
    }
    float rsum[8];
#pragma unroll
    for (int r = 0; r < 8; ++r) rsum[r] = 0.0f;
#pragma unroll
    for (int j = 0; j < 2; ++j)
#pragma unroll
      for (int r = 0; r < 8; ++r) {
        float p = __expf(sc[j][r] - mrow[r]);
        sc[j][r] = p;
        rsum[r] += p;
      }
#pragma unroll
    for (int r = 0; r < 8; ++r) {
#pragma unroll
      for (int m = 1; m < 16; m <<= 1)
        rsum[r] += __shfl_xor(rsum[r], m, 32);
      lrow[r] += rsum[r];
    }
#pragma unroll
    for (int j = 0; j < 2; ++j)
#pragma unroll
      for (int r = 0; r < 8; ++r)
        myPs[(r + hi * 8) * PS_STR + j * 16 + hl] = (bf16)sc[j][r];
    v16bf ap;
    {
      int kb = hi ? 8 : 0;
      FragBf f;
      f.h[0] = *(const v8bf*)&myPs[hl * PS_STR + kb];
      f.h[1] = *(const v8bf*)&myPs[hl * PS_STR + kb + 16];
      ap = f.v;
    }
    int kb2 = hi ? 8 : 0;
#pragma unroll
    for (int ni = 0; ni < 4; ++ni) {
      int dk = ni * 16 + hl;
      FragBf f;
      f.h[0] = *(const v8bf*)&myVs[dk * VS_STR + kb2];
      f.h[1] = *(const v8bf*)&myVs[dk * VS_STR + kb2 + 16];
      oacc[ni] = wmma_bf16(ap, f.v, oacc[ni]);
    }
  }

#pragma unroll
  for (int ni = 0; ni < 4; ++ni) {
    int dk = ni * 16 + hl;
#pragma unroll
    for (int r = 0; r < 8; ++r) {
      int q = q0 + r + hi * 8;
      float val = oacc[ni][r] / lrow[r];
      O[((size_t)(b * S + q)) * D + h * 64 + dk] = (bf16)val;
    }
  }
}

// ---------------------------------------------------------------------------
// Host-side launcher
// ---------------------------------------------------------------------------
extern "C" void kernel_launch(void* const* d_in, const int* in_sizes, int n_in,
                              void* d_out, int out_size, void* d_ws, size_t ws_size,
                              hipStream_t stream) {
  (void)in_sizes; (void)n_in; (void)out_size; (void)ws_size;
  constexpr int Bc = 2, Sc = 2048, Dc = 1024, Hc = 16;
  const int M = Bc * Sc;

  const float* x  = (const float*)d_in[0];
  const float* Wq = (const float*)d_in[2];
  const float* bq = (const float*)d_in[3];
  const float* Wk = (const float*)d_in[4];
  const float* bk = (const float*)d_in[5];
  const float* Wv = (const float*)d_in[6];
  const float* bv = (const float*)d_in[7];
  const float* Wo = (const float*)d_in[8];
  const float* bo = (const float*)d_in[9];
  float* out = (float*)d_out;

  const size_t NB = (size_t)M * Dc;        // 4.19M elements
  const size_t WW = (size_t)Dc * Dc;       // 1M elements
  bf16* xb  = (bf16*)d_ws;
  bf16* qb  = xb  + NB;
  bf16* kb_ = qb  + NB;
  bf16* vb  = kb_ + NB;
  bf16* ab  = vb  + NB;
  bf16* wtq = ab  + NB;
  bf16* wtk = wtq + WW;
  bf16* wtv = wtk + WW;
  bf16* wto = wtv + WW;                    // total ~48 MB

  int n4 = (int)(NB / 4);
  cast_f32_bf16<<<(n4 + 255) / 256, 256, 0, stream>>>(x, xb, n4);

  dim3 gt(Dc / 32, Dc / 32);               // (32, 32)
  transpose_cast<<<gt, 256, 0, stream>>>(Wq, wtq, Dc, Dc);
  transpose_cast<<<gt, 256, 0, stream>>>(Wk, wtk, Dc, Dc);
  transpose_cast<<<gt, 256, 0, stream>>>(Wv, wtv, Dc, Dc);
  transpose_cast<<<gt, 256, 0, stream>>>(Wo, wto, Dc, Dc);

  dim3 gg(Dc / 128, M / 128);              // (8, 32)
  gemm_bias_rope<<<gg, 256, 0, stream>>>(xb, wtq, bq, qb,  nullptr, M, Dc, Dc, Sc, 1);
  gemm_bias_rope<<<gg, 256, 0, stream>>>(xb, wtk, bk, kb_, nullptr, M, Dc, Dc, Sc, 1);
  gemm_bias_rope<<<gg, 256, 0, stream>>>(xb, wtv, bv, vb,  nullptr, M, Dc, Dc, Sc, 0);

  dim3 ga(Sc / 128, Hc, Bc);               // (16, 16, 2)
  attn<<<ga, 256, 0, stream>>>(qb, kb_, vb, ab, Bc, Sc, Hc, Dc);

  gemm_bias_rope<<<gg, 256, 0, stream>>>(ab, wto, bo, nullptr, out, M, Dc, Dc, Sc, 0);
}
